// LSTMModelFull_5016521801960
// MI455X (gfx1250) — compile-verified, matrix operand order checked by reading, not run on hardware
//
#include <hip/hip_runtime.h>
#include <hip/hip_bf16.h>

// ---------------------------------------------------------------------------
// Model dims (fixed by the reference)
// ---------------------------------------------------------------------------
#define NCODES 8192
#define HID    512
#define DEMO   32
#define BB     64
#define TT     50
#define KCODES 32
#define FEAT   (2 * HID + DEMO)   // 1056 (multiple of 32)
#define G4H    (4 * HID)          // 2048
#define ROWS   (TT * BB)          // 3200
#define REC_BLOCKS 16             // 8 per direction

typedef __attribute__((ext_vector_type(16))) _Float16 v16h;
typedef __attribute__((ext_vector_type(8)))  _Float16 f16x8;
typedef __attribute__((ext_vector_type(8)))  float    v8f;
typedef __attribute__((ext_vector_type(4)))  float    f32x4;
typedef __attribute__((ext_vector_type(4)))  unsigned int u32x4;
typedef __attribute__((ext_vector_type(8)))  unsigned int u32x8;

// ---------------------------------------------------------------------------
// WMMA tile helpers (CDNA5 wave32 layouts, CDNA5 ISA)
// 16-bit A/B 16x32 fragment from row-major memory with leading dim ld:
//   lane<16 : row = lane,    K = {0..7, 16..23}
//   lane>=16: row = lane-16, K = {8..15, 24..31}
// ---------------------------------------------------------------------------
__device__ __forceinline__ v16h load_tile_16x32_f16(const _Float16* p, int ld, int lane) {
    const int r  = lane & 15;
    const int kh = (lane >> 4) << 3;      // 0 or 8
    const _Float16* q = p + (long)r * ld + kh;
    f16x8 lo = *(const f16x8*)(q);        // K = kh + 0..7
    f16x8 hi = *(const f16x8*)(q + 16);   // K = kh + 16..23
    return __builtin_shufflevector(lo, hi, 0, 1, 2, 3, 4, 5, 6, 7,
                                           8, 9, 10, 11, 12, 13, 14, 15);
}

// Packed C/D fragment: each 16x16 tile stored as 256 contiguous floats,
// lane-major (lane*8 + r). Producer and consumer agree on this layout.
__device__ __forceinline__ v8f load_c_packed(const float* tile, int lane) {
    const float* q = tile + lane * 8;
    f32x4 lo = *(const f32x4*)(q);
    f32x4 hi = *(const f32x4*)(q + 4);
    return __builtin_shufflevector(lo, hi, 0, 1, 2, 3, 4, 5, 6, 7);
}

__device__ __forceinline__ float sigmoidf_(float x) {
    return 1.0f / (1.0f + __expf(-x));
}

__device__ __forceinline__ v8f wmma_f16(v16h a, v16h b, v8f c) {
    return __builtin_amdgcn_wmma_f32_16x16x32_f16(false, a, false, b,
                                                  (short)0, c, false, false);
}

// ---------------------------------------------------------------------------
// Tensor Data Mover: 1-D copy of `nelem` f16 elements from global to LDS.
// Descriptor per CDNA5 ISA (group0 128b, group1 256b,
// groups 2/3 zero). Issued as wave-uniform inline asm (portable across
// toolchains); completion via TENSORcnt.
// ---------------------------------------------------------------------------
__device__ __forceinline__ void tdm_load_1d_f16(unsigned lds_byte_off,
                                                const void* gaddr,
                                                unsigned nelem) {
    unsigned long long ga = (unsigned long long)gaddr;
    u32x4 g0;
    g0[0] = 1u;                                   // count=1 valid descriptor
    g0[1] = lds_byte_off;                         // lds_addr
    g0[2] = (unsigned)(ga & 0xFFFFFFFFu);         // global_addr[31:0]
    g0[3] = (unsigned)((ga >> 32) & 0x01FFFFFFu)  // global_addr[56:32]
            | (2u << 30);                         // type = 2 ("image")
    u32x8 g1;
    g1[0] = (1u << 16);                           // data_size=1 (2 bytes/elem)
    g1[1] = (nelem & 0xFFFFu) << 16;              // tensor_dim0[15:0]
    g1[2] = ((nelem >> 16) & 0xFFFFu) | (1u << 16); // dim0[31:16] | tensor_dim1=1
    g1[3] = (nelem & 0xFFFFu) << 16;              // tile_dim0 = nelem (<=65535)
    g1[4] = 1u;                                   // tile_dim1=1, tile_dim2=0
    g1[5] = nelem;                                // tensor_dim0_stride lo
    g1[6] = 0u;
    g1[7] = 0u;
    u32x4 gz = {0u, 0u, 0u, 0u};
    asm volatile("tensor_load_to_lds %0, %1, %2, %3"
                 :: "s"(g0), "s"(g1), "s"(gz), "s"(gz)
                 : "memory");
}

// ---------------------------------------------------------------------------
// 1) f32 -> f16 convert (grid-stride)
// ---------------------------------------------------------------------------
__global__ void cvt_f32_f16_kernel(const float* __restrict__ in,
                                   _Float16* __restrict__ out, long n) {
    long i = (long)blockIdx.x * blockDim.x + threadIdx.x;
    long s = (long)gridDim.x * blockDim.x;
    for (; i < n; i += s) out[i] = (_Float16)in[i];
}

// Wc [HID][NCODES] f32 -> WcT [NCODES][HID] f16
__global__ void transpose_wc_kernel(const float* __restrict__ Wc,
                                    _Float16* __restrict__ WcT) {
    long n = (long)NCODES * HID;
    long i = (long)blockIdx.x * blockDim.x + threadIdx.x;
    long s = (long)gridDim.x * blockDim.x;
    for (; i < n; i += s) {
        int c = (int)(i / HID);
        int h = (int)(i % HID);
        WcT[i] = (_Float16)Wc[(long)h * NCODES + c];
    }
}

// ---------------------------------------------------------------------------
// 2) Multi-hot embedding with set semantics (duplicates collapse to 1):
//    X[t*B+b][h] = bc[h] + sum over unique codes of WcT[code][h]
// ---------------------------------------------------------------------------
__global__ void embed_kernel(const int* __restrict__ codes,
                             const _Float16* __restrict__ WcT,
                             const float* __restrict__ bc,
                             _Float16* __restrict__ X) {
    const int bt = blockIdx.x;        // b*T + t
    const int b = bt / TT, t = bt % TT;
    __shared__ int s_code[KCODES];
    __shared__ int s_keep[KCODES];
    if (threadIdx.x < KCODES)
        s_code[threadIdx.x] = codes[((long)b * TT + t) * KCODES + threadIdx.x];
    __syncthreads();
    if (threadIdx.x < KCODES) {
        int keep = 1, c = s_code[threadIdx.x];
        for (int j = 0; j < (int)threadIdx.x; ++j)
            if (s_code[j] == c) keep = 0;
        s_keep[threadIdx.x] = keep;
    }
    __syncthreads();
    for (int h = threadIdx.x; h < HID; h += blockDim.x) {
        float acc = bc[h];
        for (int k = 0; k < KCODES; ++k)
            if (s_keep[k]) acc += (float)WcT[(long)s_code[k] * HID + h];
        X[((long)t * BB + b) * HID + h] = (_Float16)acc;
    }
}

// ---------------------------------------------------------------------------
// 3) WMMA GEMM: C[z] = A @ W[z]^T + bias[z];  A f16 [M x K] (lda == K),
//    W f16 row-major [N x K]. 128 threads / 4 waves; each wave computes
//    4 N-tiles (A fragment reused 4x). A tile staged in LDS by the TDM.
//    K-loop software-pipelined. Optional packed-fragment output for G.
// ---------------------------------------------------------------------------
template <typename OutT, bool RELU, bool PACKED>
__global__ __launch_bounds__(128) void gemm_tn_kernel(
    const _Float16* __restrict__ A,
    const _Float16* __restrict__ W, int ldw, long w_zs,
    const float* __restrict__ bias, long b_zs,
    OutT* __restrict__ C, int ldc, long c_zs,
    int K, int Ntiles, int Mtiles) {
    extern __shared__ char smem[];
    _Float16* aLds = (_Float16*)smem;
    const int lane = threadIdx.x & 31;
    const int wave = threadIdx.x >> 5;
    const int mi = blockIdx.x;
    const int m0 = mi * 16;
    const int nbase = (blockIdx.y * 4 + wave) * 64;   // this wave: 4 tiles
    const int z = blockIdx.z;

    const _Float16* Wz = W + (long)z * w_zs;
    const float* biasz = bias + (long)z * b_zs;

    // Stage A tile (16 contiguous rows of K halves) into LDS via TDM.
    if (wave == 0) {
        tdm_load_1d_f16(__builtin_amdgcn_groupstaticsize(),
                        A + (long)m0 * K, (unsigned)(16 * K));
        __builtin_amdgcn_s_wait_tensorcnt(0);
    }
    __syncthreads();

    v8f acc[4] = {};
    const _Float16* bp = Wz + (long)nbase * ldw;

    v16h a_c = load_tile_16x32_f16(aLds, K, lane);
    v16h b_c[4];
#pragma unroll
    for (int j = 0; j < 4; ++j)
        b_c[j] = load_tile_16x32_f16(bp + (long)j * 16 * ldw, ldw, lane);

    int kk = 0;
    for (; kk + 32 < K; kk += 32) {
        __builtin_prefetch(bp + kk + 512, 0, 1);
        v16h a_n = load_tile_16x32_f16(aLds + kk + 32, K, lane);
        v16h b_n[4];
#pragma unroll
        for (int j = 0; j < 4; ++j)
            b_n[j] = load_tile_16x32_f16(bp + (long)j * 16 * ldw + kk + 32, ldw, lane);
#pragma unroll
        for (int j = 0; j < 4; ++j) acc[j] = wmma_f16(a_c, b_c[j], acc[j]);
        a_c = a_n;
#pragma unroll
        for (int j = 0; j < 4; ++j) b_c[j] = b_n[j];
    }
#pragma unroll
    for (int j = 0; j < 4; ++j) acc[j] = wmma_f16(a_c, b_c[j], acc[j]);

    const int col = lane & 15;
    const int r0  = (lane >> 4) << 3;
#pragma unroll
    for (int j = 0; j < 4; ++j) {
        const int n0 = nbase + j * 16;
        const float bv = biasz[n0 + col];
        if constexpr (PACKED) {
            // one 16x16 tile = 256 contiguous floats, lane-major
            const long idx = ((long)z * Mtiles + mi) * Ntiles + (n0 >> 4);
            float* q = (float*)C + idx * 256 + lane * 8;
            f32x4 lo = {acc[j][0] + bv, acc[j][1] + bv, acc[j][2] + bv, acc[j][3] + bv};
            f32x4 hi = {acc[j][4] + bv, acc[j][5] + bv, acc[j][6] + bv, acc[j][7] + bv};
            *(f32x4*)(q)     = lo;
            *(f32x4*)(q + 4) = hi;
        } else {
#pragma unroll
            for (int r = 0; r < 8; ++r) {
                float v = acc[j][r] + bv;
                if (RELU) v = v > 0.0f ? v : 0.0f;
                C[(long)(m0 + r0 + r) * ldc + n0 + col] = (OutT)v;
            }
        }
    }
}

// ---------------------------------------------------------------------------
// 4) Recurrence init: zero double-buffered h state and arrival counter.
// ---------------------------------------------------------------------------
__global__ void rec_init_kernel(_Float16* __restrict__ hglob,
                                unsigned int* __restrict__ bar) {
    long n = 2L * 2 * BB * HID;
    long i = (long)blockIdx.x * blockDim.x + threadIdx.x;
    long s = (long)gridDim.x * blockDim.x;
    for (; i < n; i += s) hglob[i] = (_Float16)0.0f;
    if (blockIdx.x == 0 && threadIdx.x == 0) *bar = 0u;
}

// ---------------------------------------------------------------------------
// 5) LSTM recurrence, distributed: 16 workgroups (8 per direction), each owns
//    64 H-columns (weight streaming spread over 8 WGPs instead of 1).
//    Per step: h staged global->LDS by the TDM; gates = packed-G + h@w_hh^T
//    via WMMA (k-outer, B fragments reused across all M-tiles, software-
//    pipelined); c lives in registers; timestep sync = device-scope atomic
//    arrival counter with monotonic target (all 16 blocks co-resident).
// ---------------------------------------------------------------------------
__global__ __launch_bounds__(256) void lstm_rec_kernel(
    const float* __restrict__ Gp,      // packed [2][ROWS/16][128] tiles
    const _Float16* __restrict__ whh,  // [2][4H][H] f16
    _Float16* __restrict__ hglob,      // [2 parity][2 dir][64*512] f16
    unsigned int* __restrict__ bar,
    _Float16* __restrict__ Y,          // layer0 out: [ROWS][2H] f16
    _Float16* __restrict__ feat,       // layer1 out: [B][FEAT] f16
    int layer) {
    extern __shared__ char smem[];
    _Float16* h_lds = (_Float16*)smem;              // 64 x 512 f16 (64KB)

    const int d    = blockIdx.y;
    const int blk  = blockIdx.x;                    // 0..7
    const int tid  = threadIdx.x;
    const int lane = tid & 31;
    const int wave = tid >> 5;                      // 0..7
    const int ci   = wave & 3;                      // H-chunk within block
    const int half = wave >> 2;                     // row half (0/1)
    const int chunk = blk * 64 + ci * 16;           // H-column base
    const int col = lane & 15;
    const int r0  = (lane >> 4) << 3;

    const _Float16* whh_d = whh + (long)d * G4H * HID;

    float creg[2][8];
#pragma unroll
    for (int m = 0; m < 2; ++m)
#pragma unroll
        for (int r = 0; r < 8; ++r) creg[m][r] = 0.0f;

    for (int step = 0; step < TT; ++step) {
        const int t = d ? (TT - 1 - step) : step;
        const int p = step & 1;

        // stage current h for this direction into LDS (TDM, 64KB 1-D copy)
        if (wave == 0) {
            tdm_load_1d_f16(__builtin_amdgcn_groupstaticsize(),
                            hglob + ((long)p * 2 + d) * (BB * HID),
                            (unsigned)(BB * HID));
            __builtin_amdgcn_s_wait_tensorcnt(0);
        }
        __syncthreads();

        // C = packed pre-activations G[t]
        v8f acc[2][4];
#pragma unroll
        for (int m = 0; m < 2; ++m)
#pragma unroll
            for (int g = 0; g < 4; ++g) {
                const long idx = ((long)d * (ROWS / 16) + (t * 4 + half * 2 + m))
                                     * (G4H / 16) + (g * 32 + blk * 4 + ci);
                acc[m][g] = load_c_packed(Gp + idx * 256, lane);
            }

        // K loop: B fragments reused across both M-tiles, pipelined
        v16h b_c[4], b_n[4];
#pragma unroll
        for (int g = 0; g < 4; ++g)
            b_c[g] = load_tile_16x32_f16(whh_d + (long)(g * HID + chunk) * HID,
                                         HID, lane);
#pragma unroll
        for (int k = 0; k < 16; ++k) {
            if (k < 15) {
#pragma unroll
                for (int g = 0; g < 4; ++g)
                    b_n[g] = load_tile_16x32_f16(
                        whh_d + (long)(g * HID + chunk) * HID + (k + 1) * 32,
                        HID, lane);
            }
            v16h a0 = load_tile_16x32_f16(h_lds + (half * 32) * HID + k * 32,
                                          HID, lane);
            v16h a1 = load_tile_16x32_f16(h_lds + (half * 32 + 16) * HID + k * 32,
                                          HID, lane);
#pragma unroll
            for (int g = 0; g < 4; ++g) acc[0][g] = wmma_f16(a0, b_c[g], acc[0][g]);
#pragma unroll
            for (int g = 0; g < 4; ++g) acc[1][g] = wmma_f16(a1, b_c[g], acc[1][g]);
            if (k < 15) {
#pragma unroll
                for (int g = 0; g < 4; ++g) b_c[g] = b_n[g];
            }
        }

        // elementwise LSTM cell; write h into the other parity buffer
        _Float16* hnext = hglob + ((long)(p ^ 1) * 2 + d) * (BB * HID);
#pragma unroll
        for (int m = 0; m < 2; ++m) {
#pragma unroll
            for (int r = 0; r < 8; ++r) {
                const int row = (half * 2 + m) * 16 + r0 + r;
                const float iv = sigmoidf_(acc[m][0][r]);
                const float fv = sigmoidf_(acc[m][1][r]);
                const float gv = tanhf(acc[m][2][r]);
                const float ov = sigmoidf_(acc[m][3][r]);
                const float cnew = fv * creg[m][r] + iv * gv;
                creg[m][r] = cnew;
                const float hvf = ov * tanhf(cnew);
                const _Float16 hv = (_Float16)hvf;
                hnext[row * HID + chunk + col] = hv;
                if (layer == 0)
                    Y[((long)t * BB + row) * (2 * HID) + d * HID + chunk + col] = hv;
                if (layer == 1 && step == TT - 1)
                    feat[(long)row * FEAT + d * HID + chunk + col] = hv;
            }
        }

        // device-scope timestep barrier (16 co-resident blocks, monotonic)
        if (step < TT - 1) {
            __threadfence();
            __syncthreads();
            if (tid == 0) {
                __hip_atomic_fetch_add(bar, 1u, __ATOMIC_ACQ_REL,
                                       __HIP_MEMORY_SCOPE_AGENT);
                const unsigned target = (unsigned)REC_BLOCKS * (step + 1);
                while (__hip_atomic_load(bar, __ATOMIC_ACQUIRE,
                                         __HIP_MEMORY_SCOPE_AGENT) < target)
                    __builtin_amdgcn_s_sleep(1);
            }
            __syncthreads();
        }
    }
}

// ---------------------------------------------------------------------------
// 6) feat[:, 1024:1056] = demographic (f32 -> f16)
// ---------------------------------------------------------------------------
__global__ void demo_fill_kernel(const float* __restrict__ demo,
                                 _Float16* __restrict__ feat) {
    int i = blockIdx.x * blockDim.x + threadIdx.x;
    if (i < BB * DEMO) {
        int b = i / DEMO, j = i % DEMO;
        feat[(long)b * FEAT + 2 * HID + j] = (_Float16)demo[i];
    }
}

// ---------------------------------------------------------------------------
// Host-side launcher
// ---------------------------------------------------------------------------
extern "C" void kernel_launch(void* const* d_in, const int* in_sizes, int n_in,
                              void* d_out, int out_size, void* d_ws, size_t ws_size,
                              hipStream_t stream) {
    (void)in_sizes; (void)n_in; (void)out_size; (void)ws_size;

    const int*   codes = (const int*)d_in[0];
    const float* demo  = (const float*)d_in[1];
    const float* Wc    = (const float*)d_in[2];
    const float* bc    = (const float*)d_in[3];
    const float* wih0  = (const float*)d_in[4];
    const float* whh0  = (const float*)d_in[5];
    const float* bl0   = (const float*)d_in[6];
    const float* wih1  = (const float*)d_in[7];
    const float* whh1  = (const float*)d_in[8];
    const float* bl1   = (const float*)d_in[9];
    const float* W1    = (const float*)d_in[10];
    const float* b1    = (const float*)d_in[11];
    const float* W2    = (const float*)d_in[12];
    const float* b2    = (const float*)d_in[13];
    float* out = (float*)d_out;

    // ---- workspace layout ----
    char* ws = (char*)d_ws;
    size_t off = 0;
    auto take = [&](size_t bytes) {
        char* p = ws + off;
        off += (bytes + 255) & ~(size_t)255;
        return p;
    };
    _Float16* WcT_h  = (_Float16*)take((size_t)NCODES * HID * 2);
    _Float16* wih0_h = (_Float16*)take((size_t)2 * G4H * HID * 2);
    _Float16* whh0_h = (_Float16*)take((size_t)2 * G4H * HID * 2);
    _Float16* wih1_h = (_Float16*)take((size_t)2 * G4H * 2 * HID * 2);
    _Float16* whh1_h = (_Float16*)take((size_t)2 * G4H * HID * 2);
    _Float16* W1_h   = (_Float16*)take((size_t)HID * FEAT * 2);
    _Float16* W2_h   = (_Float16*)take((size_t)NCODES * HID * 2);
    _Float16* X_h    = (_Float16*)take((size_t)ROWS * HID * 2);
    _Float16* Y0_h   = (_Float16*)take((size_t)ROWS * 2 * HID * 2);
    _Float16* feat_h = (_Float16*)take((size_t)BB * FEAT * 2);
    _Float16* hid_h  = (_Float16*)take((size_t)BB * HID * 2);
    _Float16* hglob  = (_Float16*)take((size_t)2 * 2 * BB * HID * 2);
    unsigned int* bar = (unsigned int*)take(256);
    float*    G_f    = (float*)take((size_t)2 * ROWS * G4H * 4);   // packed tiles

    // ---- 1) weight conversion (f32 -> f16), Wc transpose ----
    const int CB = 256;
    cvt_f32_f16_kernel<<<2048, CB, 0, stream>>>(wih0, wih0_h, (long)2 * G4H * HID);
    cvt_f32_f16_kernel<<<2048, CB, 0, stream>>>(whh0, whh0_h, (long)2 * G4H * HID);
    cvt_f32_f16_kernel<<<2048, CB, 0, stream>>>(wih1, wih1_h, (long)2 * G4H * 2 * HID);
    cvt_f32_f16_kernel<<<2048, CB, 0, stream>>>(whh1, whh1_h, (long)2 * G4H * HID);
    cvt_f32_f16_kernel<<<1024, CB, 0, stream>>>(W1, W1_h, (long)HID * FEAT);
    cvt_f32_f16_kernel<<<2048, CB, 0, stream>>>(W2, W2_h, (long)NCODES * HID);
    transpose_wc_kernel<<<2048, CB, 0, stream>>>(Wc, WcT_h);

    // ---- 2) embedding: X [ROWS x HID] f16 ----
    embed_kernel<<<BB * TT, 256, 0, stream>>>(codes, WcT_h, bc, X_h);

    // ---- 3) layer 0 input gates (packed G): X @ w_ih0^T + bl0 ----
    {
        dim3 g(ROWS / 16, G4H / 256, 2);
        gemm_tn_kernel<float, false, true><<<g, 128, 16 * HID * 2, stream>>>(
            X_h, wih0_h, HID, (long)G4H * HID, bl0, G4H,
            G_f, 0, 0, HID, G4H / 16, ROWS / 16);
    }

    // ---- 4) layer 0 recurrence -> Y0 [ROWS x 1024] ----
    rec_init_kernel<<<128, 256, 0, stream>>>(hglob, bar);
    {
        dim3 g(8, 2, 1);
        lstm_rec_kernel<<<g, 256, BB * HID * 2, stream>>>(
            G_f, whh0_h, hglob, bar, Y0_h, feat_h, 0);
    }

    // ---- 5) layer 1 input gates (packed G): Y0 @ w_ih1^T + bl1 ----
    {
        dim3 g(ROWS / 16, G4H / 256, 2);
        gemm_tn_kernel<float, false, true><<<g, 128, 16 * 2 * HID * 2, stream>>>(
            Y0_h, wih1_h, 2 * HID, (long)G4H * 2 * HID, bl1, G4H,
            G_f, 0, 0, 2 * HID, G4H / 16, ROWS / 16);
    }

    // ---- 6) layer 1 recurrence -> feat[:, 0:1024] ----
    rec_init_kernel<<<128, 256, 0, stream>>>(hglob, bar);
    {
        dim3 g(8, 2, 1);
        lstm_rec_kernel<<<g, 256, BB * HID * 2, stream>>>(
            G_f, whh1_h, hglob, bar, Y0_h, feat_h, 1);
    }

    // ---- 7) feat[:, 1024:1056] = demographic ----
    demo_fill_kernel<<<(BB * DEMO + 255) / 256, 256, 0, stream>>>(demo, feat_h);

    // ---- 8) hid = relu(feat @ W1^T + b1) ----
    {
        dim3 g(BB / 16, HID / 256, 1);
        gemm_tn_kernel<_Float16, true, false><<<g, 128, 16 * FEAT * 2, stream>>>(
            feat_h, W1_h, FEAT, 0, b1, 0, hid_h, HID, 0, FEAT, 0, 0);
    }

    // ---- 9) logits = hid @ W2^T + b2 (f32, row-major) ----
    {
        dim3 g(BB / 16, NCODES / 256, 1);
        gemm_tn_kernel<float, false, false><<<g, 128, 16 * HID * 2, stream>>>(
            hid_h, W2_h, HID, 0, b2, 0, out, NCODES, 0, HID, 0, 0);
    }
}
